// MCC_45509473468992
// MI455X (gfx1250) — matrix operand order repaired and checked by reading, not compiled, not run on hardware
//
#include <hip/hip_runtime.h>
#include <hip/hip_bf16.h>
#include <math.h>

// ---------------------------------------------------------------------------
// GraphConv + dense_mincut_pool pipeline for MI455X (gfx1250, wave32, WMMA)
// Dominant work: two dense GEMMs against A (8192x8192) via
// v_wmma_f32_16x16x32_f16 (A entries are exactly 0/1 -> exact in f16).
// Row sums of A (for dinv) are folded into GEMM1's f32 staging path, and the
// big GEMM uses double-buffered LDS (1 barrier / K-step, loads overlap WMMA).
// ---------------------------------------------------------------------------

#define NN   8192
#define TDIM 256
#define FDIM 128
#define CDIM 32
#define EPSL 1e-5f

typedef _Float16 h16v __attribute__((ext_vector_type(16)));
typedef _Float16 h8v  __attribute__((ext_vector_type(8)));
typedef _Float16 h4v  __attribute__((ext_vector_type(4)));
typedef float    f8v  __attribute__((ext_vector_type(8)));
typedef float    f4v  __attribute__((ext_vector_type(4)));

// ---- workspace layout (bytes), total ~11.9 MB ----------------------------
static const size_t SCAL_OFF  = 0;                       // 16 f32 accumulators
static const size_t SS_OFF    = 256;                     // 32x32 f32 (s^T s)
static const size_t DINV_OFF  = 8192;                    // 8192 f32
static const size_t BIAS2_OFF = DINV_OFF + 32768;        // 32 f32
static const size_t S_OFF     = BIAS2_OFF + 256;         // 8192x32 f32 softmax
static const size_t XH_OFF    = S_OFF + 1048576;         // 8192x256 f16 Xn
static const size_t AGG_OFF   = XH_OFF + 4194304;        // 8192x256 f16 agg
static const size_t WC_OFF    = AGG_OFF + 4194304;       // 32x512 f16 weights
static const size_t SP_OFF    = WC_OFF + 32768;          // 8192x48 f16
static const size_t ASP_OFF   = SP_OFF + 786432;         // 8192x48 f32

// ---------------------------------------------------------------------------
__global__ void k_init(float* scal, float* ssb) {
    int t = threadIdx.x;
    if (t < 16) scal[t] = 0.f;
    for (int i = t; i < CDIM * CDIM; i += 256) ssb[i] = 0.f;
}

__global__ void k_xstats(const float* __restrict__ X, float* scal) {
    __shared__ float s1[256], s2[256];
    int t = threadIdx.x;
    float a = 0.f, b = 0.f;
    for (int i = blockIdx.x * 256 + t; i < NN * TDIM; i += gridDim.x * 256) {
        float x = X[i]; a += x; b += x * x;
    }
    s1[t] = a; s2[t] = b; __syncthreads();
    for (int o = 128; o > 0; o >>= 1) {
        if (t < o) { s1[t] += s1[t + o]; s2[t] += s2[t + o]; }
        __syncthreads();
    }
    if (t == 0) { atomicAdd(&scal[0], s1[0]); atomicAdd(&scal[1], s2[0]); }
}

__global__ void k_xnorm(const float* __restrict__ X, const float* __restrict__ scal,
                        _Float16* __restrict__ Xh) {
    int i = blockIdx.x * 256 + threadIdx.x;
    const float inv_n = 1.f / (float)(NN * TDIM);
    float mu   = scal[0] * inv_n;
    float var  = scal[1] * inv_n - mu * mu;
    float rstd = rsqrtf(var + EPSL);
    if (i < NN * TDIM) Xh[i] = (_Float16)((X[i] - mu) * rstd);
}

// Wc[c][k<256] = (W_mlp @ W_rel)[c][k]; Wc[c][256+k] = (W_mlp @ W_root)[c][k]
// bias2[c] = (W_mlp @ b_rel)[c] + b_mlp[c]
__global__ void k_wc(const float* __restrict__ W_rel, const float* __restrict__ W_root,
                     const float* __restrict__ W_mlp, const float* __restrict__ b_rel,
                     const float* __restrict__ b_mlp,
                     _Float16* __restrict__ Wc, float* __restrict__ bias2) {
    int idx = blockIdx.x * 256 + threadIdx.x;      // 0..16383
    int c = idx >> 9;
    int k = idx & 511;
    const float* Wsrc = (k < TDIM) ? W_rel : W_root;
    int kk = k & (TDIM - 1);
    float acc = 0.f;
    for (int f = 0; f < FDIM; ++f) acc += W_mlp[c * FDIM + f] * Wsrc[f * TDIM + kk];
    Wc[c * 512 + k] = (_Float16)acc;
    if (idx < CDIM) {
        float b = b_mlp[idx];
        for (int f = 0; f < FDIM; ++f) b += W_mlp[idx * FDIM + f] * b_rel[f];
        bias2[idx] = b;
    }
}

// ---------------------------------------------------------------------------
// Big WMMA GEMM, double-buffered LDS: C[128 x BN tile] = f16(A_f32) @ B_f16.
// A: f32 [*,K] row-major, converted f32->f16 during staging; when ROWSUM, the
// exact f32 row sums accumulate on the side and n0==0 blocks emit dinv.
// B: f16 [K, ldb] row-major; staged transposed so B-fragments are contiguous.
// 8 waves: wave w owns output rows [w*16, w*16+16), all BN/16 column tiles.
// ---------------------------------------------------------------------------
template <int BN, bool OUT_F16, bool ROWSUM>
__global__ __launch_bounds__(256, 1) void k_gemm_a32(
    const float* __restrict__ A, const _Float16* __restrict__ B,
    void* __restrict__ Cout, float* __restrict__ dinv,
    int K, int ldb, int ldc) {
    constexpr int AP = 40;   // padded LDS pitch (halfs) -> conflict-free b128 reads
    constexpr int BP = 40;
    constexpr int NT = BN / 16;
    constexpr int BCH = (32 * BN) / 4;         // total 4-half B chunks per tile
    __shared__ _Float16 As[2][128 * AP];
    __shared__ _Float16 Bs[2][BN * BP];
    __shared__ float rs[128];

    const int tid  = threadIdx.x;
    const int lane = tid & 31;
    const int wave = tid >> 5;
    const int m0   = blockIdx.x * 128;
    const int n0   = blockIdx.y * BN;

    const f8v zero8 = {0.f, 0.f, 0.f, 0.f, 0.f, 0.f, 0.f, 0.f};
    f8v acc[NT];
#pragma unroll
    for (int i = 0; i < NT; ++i) acc[i] = zero8;

    // ISA 16-bit A-frag layout: lanes<16: K=akb..akb+7 then K=akb+16..akb+23
    const int arow = wave * 16 + (lane & 15);
    const int akb  = (lane >> 4) * 8;
    // ISA 16-bit B-frag layout: lanes<16: K=0..7,16..23, lanes>=16: K=8..15,24..31 (N=lane&15)
    const int bkk  = (lane >> 4) * 8;

    // staging coordinates: A chunk j -> row ar0+j*32, cols asg..asg+3 (f32)
    const int ar0 = tid >> 3;
    const int asg = (tid & 7) * 4;
    float rsum[4] = {0.f, 0.f, 0.f, 0.f};

    f4v av[4];
    h4v bv[2];

    auto loadT = [&](int k0) {
#pragma unroll
        for (int j = 0; j < 4; ++j)
            av[j] = *(const f4v*)(A + (size_t)(m0 + ar0 + j * 32) * K + k0 + asg);
#pragma unroll
        for (int i = 0; i < 2; ++i) {
            int c = tid + i * 256;
            if (c < BCH) {
                int k = c / (BN / 4), q = (c % (BN / 4)) * 4;
                bv[i] = *(const h4v*)(B + (size_t)(k0 + k) * ldb + n0 + q);
            }
        }
    };
    auto storeT = [&](int p) {
#pragma unroll
        for (int j = 0; j < 4; ++j) {
            f4v v = av[j];
            if (ROWSUM) rsum[j] += v.x + v.y + v.z + v.w;
            h4v h = {(_Float16)v.x, (_Float16)v.y, (_Float16)v.z, (_Float16)v.w};
            *(h4v*)&As[p][(ar0 + j * 32) * AP + asg] = h;
        }
#pragma unroll
        for (int i = 0; i < 2; ++i) {
            int c = tid + i * 256;
            if (c < BCH) {
                int k = c / (BN / 4), q = (c % (BN / 4)) * 4;
                Bs[p][(q + 0) * BP + k] = bv[i][0];
                Bs[p][(q + 1) * BP + k] = bv[i][1];
                Bs[p][(q + 2) * BP + k] = bv[i][2];
                Bs[p][(q + 3) * BP + k] = bv[i][3];
            }
        }
    };

    // prologue: stage tile 0
    loadT(0);
    storeT(0);
    __syncthreads();

    int p = 0;
    for (int k0 = 0; k0 < K; k0 += 32) {
        const bool more = (k0 + 32) < K;
        if (more) loadT(k0 + 32);        // global loads overlap this tile's WMMAs

        h8v alo = *(const h8v*)&As[p][arow * AP + akb];
        h8v ahi = *(const h8v*)&As[p][arow * AP + akb + 16];
        h16v af = __builtin_shufflevector(alo, ahi, 0, 1, 2, 3, 4, 5, 6, 7,
                                          8, 9, 10, 11, 12, 13, 14, 15);
#pragma unroll
        for (int nt = 0; nt < NT; ++nt) {
            int ncol = nt * 16 + (lane & 15);
            h8v blo = *(const h8v*)&Bs[p][ncol * BP + bkk];
            h8v bhi = *(const h8v*)&Bs[p][ncol * BP + bkk + 16];
            h16v bf = __builtin_shufflevector(blo, bhi, 0, 1, 2, 3, 4, 5, 6, 7,
                                              8, 9, 10, 11, 12, 13, 14, 15);
            acc[nt] = __builtin_amdgcn_wmma_f32_16x16x32_f16(
                false, af, false, bf, (short)0, acc[nt], false, false);
        }
        if (more) {
            storeT(p ^ 1);               // fill the other buffer
            __syncthreads();             // single barrier per K-step
            p ^= 1;
        }
    }

    // ---- epilogue: C f32 16x16 layout (VGPR v -> M=v / v+8) ----------------
    const int orow = m0 + wave * 16 + ((lane >> 4) ? 8 : 0);
    const int ocl  = lane & 15;
#pragma unroll
    for (int nt = 0; nt < NT; ++nt) {
        int col = n0 + nt * 16 + ocl;
#pragma unroll
        for (int v = 0; v < 8; ++v) {
            if constexpr (OUT_F16)
                ((_Float16*)Cout)[(size_t)(orow + v) * ldc + col] = (_Float16)acc[nt][v];
            else
                ((float*)Cout)[(size_t)(orow + v) * ldc + col] = acc[nt][v];
        }
    }

    // ---- folded row-sum reduction: dinv[m0+r] = rsqrt(sum_j A[m0+r][j]) ----
    if constexpr (ROWSUM) {
        __syncthreads();
        if (tid < 128) rs[tid] = 0.f;
        __syncthreads();
#pragma unroll
        for (int j = 0; j < 4; ++j) atomicAdd(&rs[ar0 + j * 32], rsum[j]);
        __syncthreads();
        if (n0 == 0 && tid < 128) dinv[m0 + tid] = rsqrtf(rs[tid]);
    }
}

// ---------------------------------------------------------------------------
// S = [agg | Xn] (f16, K=512) @ Wc^T + bias2   -> f32 [8192 x 32] into d_out
// ---------------------------------------------------------------------------
__global__ __launch_bounds__(256, 1) void k_gemm_s(
    const _Float16* __restrict__ AGG, const _Float16* __restrict__ XH,
    const _Float16* __restrict__ WC, const float* __restrict__ bias2,
    float* __restrict__ S) {
    constexpr int AP = 40, BP = 40;
    __shared__ _Float16 As[128 * AP];
    __shared__ _Float16 Bs[32 * BP];
    const int tid = threadIdx.x, lane = tid & 31, wave = tid >> 5;
    const int m0 = blockIdx.x * 128;

    const f8v zero8 = {0.f, 0.f, 0.f, 0.f, 0.f, 0.f, 0.f, 0.f};
    f8v acc[2] = {zero8, zero8};

    const int arow = wave * 16 + (lane & 15);
    const int akb  = (lane >> 4) * 8;
    const int bkk  = (lane >> 4) * 8;

    for (int k0 = 0; k0 < 512; k0 += 32) {
        const _Float16* src = (k0 < 256) ? (AGG + k0) : (XH + (k0 - 256));
#pragma unroll
        for (int j = 0; j < 4; ++j) {
            int c = tid + j * 256;
            int r = c >> 3, sg = (c & 7) * 4;
            *(h4v*)&As[r * AP + sg] = *(const h4v*)(src + (size_t)(m0 + r) * 256 + sg);
        }
        {   // Bs[n][k] = Wc[n][k0+k] : contiguous row-segment copies
            int n = tid >> 3, q = (tid & 7) * 4;
            *(h4v*)&Bs[n * BP + q] = *(const h4v*)(WC + n * 512 + k0 + q);
        }
        __syncthreads();

        h8v alo = *(const h8v*)&As[arow * AP + akb];
        h8v ahi = *(const h8v*)&As[arow * AP + akb + 16];
        h16v af = __builtin_shufflevector(alo, ahi, 0, 1, 2, 3, 4, 5, 6, 7,
                                          8, 9, 10, 11, 12, 13, 14, 15);
#pragma unroll
        for (int nt = 0; nt < 2; ++nt) {
            int ncol = nt * 16 + (lane & 15);
            h8v blo = *(const h8v*)&Bs[ncol * BP + bkk];
            h8v bhi = *(const h8v*)&Bs[ncol * BP + bkk + 16];
            h16v bf = __builtin_shufflevector(blo, bhi, 0, 1, 2, 3, 4, 5, 6, 7,
                                              8, 9, 10, 11, 12, 13, 14, 15);
            acc[nt] = __builtin_amdgcn_wmma_f32_16x16x32_f16(
                false, af, false, bf, (short)0, acc[nt], false, false);
        }
        __syncthreads();
    }

    const int orow = m0 + wave * 16 + ((lane >> 4) ? 8 : 0);
    const int ocl  = lane & 15;
#pragma unroll
    for (int nt = 0; nt < 2; ++nt) {
        int col = nt * 16 + ocl;
        float b = bias2[col];
#pragma unroll
        for (int v = 0; v < 8; ++v)
            S[(size_t)(orow + v) * CDIM + col] = acc[nt][v] + b;
    }
}

// ---------------------------------------------------------------------------
// fused: row softmax -> s (f32), sp = [dinv*s | dinv | 0] (f16, 48 cols),
// plus this 256-row block's partial of ss = s^T s (atomic into ssb).
// ---------------------------------------------------------------------------
__global__ __launch_bounds__(256, 1) void k_softmax_ss(
    const float* __restrict__ S, const float* __restrict__ dinv,
    float* __restrict__ sout, _Float16* __restrict__ sp, float* __restrict__ ssb) {
    __shared__ float sh[256 * CDIM];
    int t = threadIdx.x;
    int i = blockIdx.x * 256 + t;

    float v[CDIM];
    float m = -1e30f;
#pragma unroll
    for (int c = 0; c < CDIM; ++c) { v[c] = S[i * CDIM + c]; m = fmaxf(m, v[c]); }
    float sum = 0.f;
#pragma unroll
    for (int c = 0; c < CDIM; ++c) { v[c] = __expf(v[c] - m); sum += v[c]; }
    float inv = 1.f / sum;
    float di  = dinv[i];
#pragma unroll
    for (int c = 0; c < CDIM; ++c) {
        float s = v[c] * inv;
        sout[i * CDIM + c] = s;
        sh[t * CDIM + c]   = s;
        sp[i * 48 + c]     = (_Float16)(s * di);
    }
    sp[i * 48 + 32] = (_Float16)di;
#pragma unroll
    for (int c = 33; c < 48; ++c) sp[i * 48 + c] = (_Float16)0.f;
    __syncthreads();

    for (int pidx = t; pidx < CDIM * CDIM; pidx += 256) {
        int c = pidx >> 5, c2 = pidx & 31;
        float acc = 0.f;
        for (int r = 0; r < 256; ++r) acc += sh[r * CDIM + c] * sh[r * CDIM + c2];
        atomicAdd(&ssb[pidx], acc);
    }
}

// mincut_num += sum_c s*dinv*Asp[:,c];  mincut_den += dinv*Asp[:,32]*sum_c s^2
__global__ void k_mincut(const float* __restrict__ s, const float* __restrict__ Asp,
                         const float* __restrict__ dinv, float* scal) {
    __shared__ float sn[256], sd[256];
    int t = threadIdx.x;
    int i = blockIdx.x * 256 + t;
    float num = 0.f, den = 0.f;
    {
        float di = dinv[i], ssq = 0.f;
#pragma unroll
        for (int c = 0; c < CDIM; ++c) {
            float sv = s[i * CDIM + c];
            num += sv * di * Asp[i * 48 + c];
            ssq += sv * sv;
        }
        den = di * Asp[i * 48 + 32] * ssq;
    }
    sn[t] = num; sd[t] = den; __syncthreads();
    for (int o = 128; o > 0; o >>= 1) {
        if (t < o) { sn[t] += sn[t + o]; sd[t] += sd[t + o]; }
        __syncthreads();
    }
    if (t == 0) { atomicAdd(&scal[2], sn[0]); atomicAdd(&scal[3], sd[0]); }
}

__global__ void k_final(const float* __restrict__ ssb, const float* __restrict__ scal,
                        float* __restrict__ out) {
    __shared__ float red[256];
    __shared__ float nrm_s;
    int t = threadIdx.x;
    float a = 0.f;
    for (int p = t; p < CDIM * CDIM; p += 256) { float v = ssb[p]; a += v * v; }
    red[t] = a; __syncthreads();
    for (int o = 128; o > 0; o >>= 1) { if (t < o) red[t] += red[t + o]; __syncthreads(); }
    if (t == 0) nrm_s = sqrtf(red[0]);
    __syncthreads();
    float nrm = nrm_s;
    const float isq = rsqrtf((float)CDIM);
    float b = 0.f;
    for (int p = t; p < CDIM * CDIM; p += 256) {
        int c = p >> 5, c2 = p & 31;
        float v = ssb[p] / nrm - ((c == c2) ? isq : 0.f);
        b += v * v;
    }
    red[t] = b; __syncthreads();
    for (int o = 128; o > 0; o >>= 1) { if (t < o) red[t] += red[t + o]; __syncthreads(); }
    if (t == 0) {
        out[0] = -(scal[2] / scal[3]);   // loss_mc
        out[1] = sqrtf(red[0]);          // loss_o
    }
}

// ---------------------------------------------------------------------------
extern "C" void kernel_launch(void* const* d_in, const int* in_sizes, int n_in,
                              void* d_out, int out_size, void* d_ws, size_t ws_size,
                              hipStream_t stream) {
    const float* X      = (const float*)d_in[0];
    const float* A      = (const float*)d_in[1];
    const float* W_rel  = (const float*)d_in[2];
    const float* b_rel  = (const float*)d_in[3];
    const float* W_root = (const float*)d_in[4];
    const float* W_mlp  = (const float*)d_in[5];
    const float* b_mlp  = (const float*)d_in[6];

    char* ws = (char*)d_ws;
    float*    scal  = (float*)(ws + SCAL_OFF);
    float*    ssb   = (float*)(ws + SS_OFF);
    float*    dinv  = (float*)(ws + DINV_OFF);
    float*    bias2 = (float*)(ws + BIAS2_OFF);
    float*    sbuf  = (float*)(ws + S_OFF);
    _Float16* Xh    = (_Float16*)(ws + XH_OFF);
    _Float16* agg   = (_Float16*)(ws + AGG_OFF);
    _Float16* Wc    = (_Float16*)(ws + WC_OFF);
    _Float16* sp    = (_Float16*)(ws + SP_OFF);
    float*    Asp   = (float*)(ws + ASP_OFF);
    float*    S     = (float*)d_out;

    k_init      <<<1, 256, 0, stream>>>(scal, ssb);
    k_xstats    <<<256, 256, 0, stream>>>(X, scal);
    k_xnorm     <<<(NN * TDIM) / 256, 256, 0, stream>>>(X, scal, Xh);
    k_wc        <<<64, 256, 0, stream>>>(W_rel, W_root, W_mlp, b_rel, b_mlp, Wc, bias2);
    // pass 1: agg = A @ Xn (f16 WMMA, f32 accum); also emits dinv from f32 row sums
    k_gemm_a32<64, true, true><<<dim3(NN / 128, TDIM / 64), 256, 0, stream>>>(
        A, Xh, agg, dinv, NN, TDIM, TDIM);
    // S = [agg | Xn] @ Wc^T + bias2
    k_gemm_s    <<<NN / 128, 256, 0, stream>>>(agg, Xh, Wc, bias2, S);
    k_softmax_ss<<<NN / 256, 256, 0, stream>>>(S, dinv, sbuf, sp, ssb);
    // pass 2: Asp = A @ [dinv*s | dinv | 0]  (48 cols, f32 out)
    k_gemm_a32<48, false, false><<<dim3(NN / 128, 1), 256, 0, stream>>>(
        A, sp, Asp, nullptr, NN, 48, 48);
    k_mincut    <<<NN / 256, 256, 0, stream>>>(sbuf, Asp, dinv, scal);
    k_final     <<<1, 256, 0, stream>>>(ssb, scal, S + (size_t)NN * CDIM);
}
